// GroupedQueryAttention_80264348828230
// MI455X (gfx1250) — matrix operand order repaired and checked by reading, not compiled, not run on hardware
//
#include <hip/hip_runtime.h>
#include <hip/hip_bf16.h>
#include <math.h>

typedef __bf16 bf16_t;
typedef __attribute__((ext_vector_type(16))) __bf16 v16bf;
typedef __attribute__((ext_vector_type(8)))  float   v8f;
typedef __attribute__((ext_vector_type(4)))  unsigned int v4u;
typedef __attribute__((ext_vector_type(8)))  int v8i;
typedef __attribute__((ext_vector_type(4)))  int v4i;

union FragA { v16bf v; unsigned int u[8]; };

constexpr int Bb  = 2;
constexpr int Nn  = 2048;
constexpr int Dd  = 2048;
constexpr int Hh  = 16;
constexpr int KVH = 4;
constexpr int DHd = 128;
constexpr int GROUP = Hh / KVH;

// ---- feature detection for the TDM path ----
#if defined(__has_builtin)
# if __has_builtin(__builtin_amdgcn_tensor_load_to_lds)
#  define HAVE_TDM 1
# else
#  define HAVE_TDM 0
# endif
# if __has_builtin(__builtin_amdgcn_s_wait_tensorcnt)
#  define HAVE_WAIT_TENSOR 1
# else
#  define HAVE_WAIT_TENSOR 0
# endif
#else
# define HAVE_TDM 0
# define HAVE_WAIT_TENSOR 0
#endif

#if defined(__has_include)
# if __has_include(<hip/amd_detail/amd_gfx1250_TDM.h>)
#  define TDM_SIX_ARGS 1
# else
#  define TDM_SIX_ARGS 0
# endif
#else
# define TDM_SIX_ARGS 0
#endif

#if HAVE_TDM
#pragma message("CDNA5 probe: tensor_load_to_lds builtin AVAILABLE -> TDM path enabled")
#else
#pragma message("CDNA5 probe: tensor_load_to_lds builtin NOT available -> manual staging fallback")
#endif
#if TDM_SIX_ARGS
#pragma message("CDNA5 probe: using 6-arg TDM builtin form (therock header present)")
#else
#pragma message("CDNA5 probe: using 5-arg TDM builtin form")
#endif
#if HAVE_WAIT_TENSOR
#pragma message("CDNA5 probe: s_wait_tensorcnt builtin AVAILABLE")
#else
#pragma message("CDNA5 probe: s_wait_tensorcnt builtin NOT available -> inline asm")
#endif

__device__ __forceinline__ v8f zero_v8f() {
  v8f z;
#pragma unroll
  for (int i = 0; i < 8; ++i) z[i] = 0.0f;
  return z;
}

__device__ __forceinline__ void wait_tensorcnt0() {
#if HAVE_WAIT_TENSOR
  __builtin_amdgcn_s_wait_tensorcnt((short)0);
#else
  __asm__ volatile("s_wait_tensorcnt 0x0" ::: "memory");
#endif
}

#if HAVE_TDM
// Issue a 2-D TDM load: tile_h rows x tile_w bf16 elements, row-major source with
// row stride row_elems (in elements), destination LDS with per-row padding.
// pad_interval_code: pad after 2^(code+1) dwords stored; pad_amount_code: (code+1) dwords.
__device__ __forceinline__ void tdm_load_2d(unsigned lds_off, const void* gaddr,
                                            unsigned row_elems, unsigned tile_w,
                                            unsigned tile_h, unsigned tensor_rows,
                                            unsigned pad_interval_code,
                                            unsigned pad_amount_code) {
  const unsigned long long ga = (unsigned long long)gaddr;
  v4u g0; v8i g1; v4i g2; v4i g3;
  g0[0] = 1u;                                                    // count=1, user D#
  g0[1] = lds_off;                                               // lds_addr
  g0[2] = (unsigned)ga;                                          // global_addr[31:0]
  g0[3] = (unsigned)((ga >> 32) & 0x01FFFFFFull) | (2u << 30);   // addr[56:32] | type=2
  // workgroup_mask=0, data_size=2B, pad_enable=1
  g1[0] = (int)((1u << 16) | (1u << 20) | (pad_interval_code << 22) | (pad_amount_code << 25));
  g1[1] = (int)((row_elems & 0xFFFFu) << 16);                                  // tensor_dim0 lo16
  g1[2] = (int)(((row_elems >> 16) & 0xFFFFu) | ((tensor_rows & 0xFFFFu) << 16)); // dim0 hi | dim1 lo
  g1[3] = (int)(((tensor_rows >> 16) & 0xFFFFu) | ((tile_w & 0xFFFFu) << 16));    // dim1 hi | tile_dim0
  g1[4] = (int)(tile_h & 0xFFFFu);                                             // tile_dim1 (tile_dim2=0)
  g1[5] = (int)row_elems;                                                      // tensor_dim0_stride
  g1[6] = 0; g1[7] = 0;
  g2[0] = 0; g2[1] = 0; g2[2] = 0; g2[3] = 0;
  g3[0] = 0; g3[1] = 0; g3[2] = 0; g3[3] = 0;
# if TDM_SIX_ARGS
  v8i g4;
#pragma unroll
  for (int i = 0; i < 8; ++i) g4[i] = 0;
  __builtin_amdgcn_tensor_load_to_lds(g0, g1, g2, g3, g4, 0);
# else
  __builtin_amdgcn_tensor_load_to_lds(g0, g1, g2, g3, 0);
# endif
}
#endif  // HAVE_TDM

// ---------------- fp32 -> bf16 convert ----------------
__global__ void cvt_bf16_kernel(const float* __restrict__ in, bf16_t* __restrict__ out, size_t n) {
  size_t i = (size_t)blockIdx.x * blockDim.x + threadIdx.x;
  size_t stride = (size_t)gridDim.x * blockDim.x;
  for (; i < n; i += stride) out[i] = (bf16_t)in[i];
}

// ---------------- generic WMMA GEMM ----------------
// C[M x N] = A[M x K](bf16 row-major) * B[K x N](bf16 row-major)
// block = 128 threads (4 waves), tile = 64x64, K-step = 32.
// B tiles are TDM double-buffered into LDS row-major [k][n] (68-elem padded rows).
template <typename OutT>
__global__ void __launch_bounds__(128)
wmma_gemm_kernel(const bf16_t* __restrict__ A, const bf16_t* __restrict__ Bm,
                 OutT* __restrict__ C, int M, int N, int K) {
  __shared__ bf16_t bS[2][32][68];     // [buf][k][n], 34-dword row stride

  const int tid  = threadIdx.x;
  const int lane = tid & 31;
  const int wave = tid >> 5;
  const int hi   = lane >> 4;
  const int ln   = lane & 15;

  const int tile_m = blockIdx.y * 64;
  const int tile_n = blockIdx.x * 64;
  const int arow   = tile_m + wave * 16 + ln;
  const int nsteps = K >> 5;

#if HAVE_TDM
  if (tid < 32)
    tdm_load_2d((unsigned)(size_t)(void*)&bS[0][0][0], Bm + tile_n,
                (unsigned)N, 64u, 32u, (unsigned)K, 4u /*32 dw*/, 1u /*2 dw pad*/);
#endif

  v8f acc[4] = {zero_v8f(), zero_v8f(), zero_v8f(), zero_v8f()};
  const bf16_t* ap_base = A + (size_t)arow * K + 8 * hi;

  for (int s = 0; s < nsteps; ++s) {
    const int k0  = s << 5;
    const int cur = s & 1;

#if HAVE_TDM
    if (tid < 32) wait_tensorcnt0();
    __syncthreads();                       // publish tile s to all waves
    if ((tid < 32) && (s + 1 < nsteps))
      tdm_load_2d((unsigned)(size_t)(void*)&bS[cur ^ 1][0][0],
                  Bm + (size_t)(k0 + 32) * N + tile_n,
                  (unsigned)N, 64u, 32u, (unsigned)(K - (k0 + 32)), 4u, 1u);
#else
    __syncthreads();
    {
      const int kk = tid >> 2;             // 0..31
      const int ng = (tid & 3) << 4;       // 0,16,32,48
      const bf16_t* src = Bm + (size_t)(k0 + kk) * N + tile_n + ng;
#pragma unroll
      for (int j = 0; j < 16; ++j) bS[cur][kk][ng + j] = src[j];
    }
    __syncthreads();
#endif

    // A fragment from global (lane = row, K interleaved per ISA layout)
    FragA af;
#pragma unroll
    for (int p = 0; p < 8; ++p) {
      const int k = ((p & 3) << 1) + ((p >> 2) << 4);
      af.u[p] = *(const unsigned int*)(ap_base + k0 + k);
    }

#pragma unroll
    for (int nt = 0; nt < 4; ++nt) {
      const int ncol = nt * 16 + ln;
      FragA bfv;
#pragma unroll
      for (int p = 0; p < 8; ++p) {
        const int krow = 8 * hi + ((p & 3) << 1) + ((p >> 2) << 4);
        const unsigned lo = *(const unsigned short*)&bS[cur][krow][ncol];
        const unsigned h2 = *(const unsigned short*)&bS[cur][krow + 1][ncol];
        bfv.u[p] = lo | (h2 << 16);
      }
      acc[nt] = __builtin_amdgcn_wmma_f32_16x16x32_bf16(
          false, af.v, false, bfv.v, (short)0, acc[nt], false, false);
    }
  }

  // store: C/D layout: elem v -> row v + 8*hi, col = ln
  const int rbase = tile_m + wave * 16 + hi * 8;
#pragma unroll
  for (int nt = 0; nt < 4; ++nt) {
    const int col = tile_n + nt * 16 + ln;
#pragma unroll
    for (int v = 0; v < 8; ++v)
      C[(size_t)(rbase + v) * N + col] = (OutT)acc[nt][v];
  }
}

// ---------------- RoPE + head relayout ----------------
// in : (B, N, nheads*DHd) bf16      out : (B, nheads, N, DHd) bf16
__global__ void rope_relayout_kernel(const bf16_t* __restrict__ in, bf16_t* __restrict__ out,
                                     int nheads, float scale, int applyRope) {
  const int blk = blockIdx.x;                // ((b*nheads)+h)*Nn + n
  const int n = blk & (Nn - 1);
  const int h = (blk >> 11) % nheads;        // Nn == 2^11
  const int b = blk / (nheads * Nn);
  const int d = threadIdx.x;                 // 0..63 (pairs d, d+64)

  const size_t ib = ((size_t)b * Nn + n) * ((size_t)nheads * DHd) + (size_t)h * DHd;
  const size_t ob = (((size_t)b * nheads + h) * Nn + n) * DHd;

  float x1 = (float)in[ib + d];
  float x2 = (float)in[ib + d + 64];
  float o1 = x1, o2 = x2;
  if (applyRope) {
    const float ang = (float)n * powf(10000.0f, -(float)(2 * d) / 128.0f);
    const float c = cosf(ang), s = sinf(ang);
    o1 = x1 * c - x2 * s;
    o2 = x2 * c + x1 * s;
  }
  out[ob + d]      = (bf16_t)(o1 * scale);
  out[ob + d + 64] = (bf16_t)(o2 * scale);
}

// ---------------- flash attention (causal, GQA) ----------------
// Q: (B*H, N, 128) scaled+roped, K/V: (B*KVH, N, 128), Ctx: (B, N, H*128)
// K and V tiles are TDM double-buffered into LDS row-major (65-dword padded rows).
__global__ void __launch_bounds__(128)
attn_kernel(const bf16_t* __restrict__ Q, const bf16_t* __restrict__ Kmat,
            const bf16_t* __restrict__ V, bf16_t* __restrict__ Ctx) {
  __shared__ bf16_t kS[2][64][130];    // [buf][key][d]
  __shared__ bf16_t vS[2][64][130];    // [buf][key][d]
  __shared__ bf16_t pS[4][16][66];     // per-wave P tile: [row][key]

  const int tid  = threadIdx.x;
  const int lane = tid & 31;
  const int wave = tid >> 5;
  const int hi   = lane >> 4;
  const int ln   = lane & 15;

  const int qt  = blockIdx.x;          // query tile (64 rows)
  const int bh  = blockIdx.y;          // b*H + h
  const int b   = bh / Hh;
  const int h   = bh % Hh;
  const int kvh = h / GROUP;

  const bf16_t* Qh = Q    + (size_t)bh * Nn * DHd;
  const bf16_t* Kh = Kmat + (size_t)(b * KVH + kvh) * Nn * DHd;
  const bf16_t* Vh = V    + (size_t)(b * KVH + kvh) * Nn * DHd;

  const int qr0 = qt * 64 + wave * 16; // wave's first query row
  const int ntiles = qt + 1;           // causal: only key tiles <= query tile

  // preload Q A-fragments for all 4 K-steps (held in VGPRs for the whole kernel)
  FragA qf[4];
  {
    const bf16_t* qrow = Qh + (size_t)(qr0 + ln) * DHd + 8 * hi;
#pragma unroll
    for (int ks = 0; ks < 4; ++ks) {
#pragma unroll
      for (int p = 0; p < 8; ++p) {
        const int k = ks * 32 + ((p & 3) << 1) + ((p >> 2) << 4);
        qf[ks].u[p] = *(const unsigned int*)(qrow + k);
      }
    }
  }

#if HAVE_TDM
  if (tid < 32) {                      // wave 0 drives the DMA pipeline
    tdm_load_2d((unsigned)(size_t)(void*)&kS[0][0][0], Kh, 128u, 128u, 64u,
                (unsigned)Nn, 5u /*64 dw*/, 0u /*1 dw pad*/);
    tdm_load_2d((unsigned)(size_t)(void*)&vS[0][0][0], Vh, 128u, 128u, 64u,
                (unsigned)Nn, 5u, 0u);
  }
#endif

  float mrow[8], lrow[8];
#pragma unroll
  for (int v = 0; v < 8; ++v) { mrow[v] = -1e30f; lrow[v] = 0.0f; }
  v8f oacc[8];
#pragma unroll
  for (int t = 0; t < 8; ++t) oacc[t] = zero_v8f();

  for (int kb = 0; kb < ntiles; ++kb) {
    const int key0 = kb * 64;
    const int cur  = kb & 1;

#if HAVE_TDM
    if (tid < 32) wait_tensorcnt0();   // tile kb has landed
    __syncthreads();                   // publish to all waves
    if ((tid < 32) && (kb + 1 < ntiles)) {
      const int nk0 = (kb + 1) * 64;   // overlap next tile's DMA with this tile's math
      tdm_load_2d((unsigned)(size_t)(void*)&kS[cur ^ 1][0][0],
                  Kh + (size_t)nk0 * DHd, 128u, 128u, 64u, (unsigned)(Nn - nk0), 5u, 0u);
      tdm_load_2d((unsigned)(size_t)(void*)&vS[cur ^ 1][0][0],
                  Vh + (size_t)nk0 * DHd, 128u, 128u, 64u, (unsigned)(Nn - nk0), 5u, 0u);
    }
#else
    __syncthreads();
    {
      const int kk    = tid & 63;
      const int dbase = (tid >> 6) << 6;
      const bf16_t* ksrc = Kh + (size_t)(key0 + kk) * DHd + dbase;
      const bf16_t* vsrc = Vh + (size_t)(key0 + kk) * DHd + dbase;
#pragma unroll 8
      for (int j = 0; j < 64; ++j) kS[cur][kk][dbase + j] = ksrc[j];
#pragma unroll 8
      for (int j = 0; j < 64; ++j) vS[cur][kk][dbase + j] = vsrc[j];
    }
    __syncthreads();
#endif

    // ---- S = Q * K^T for 64 keys (B-frags from LDS, K rows contiguous) ----
    v8f sacc[4];
#pragma unroll
    for (int nt = 0; nt < 4; ++nt) sacc[nt] = zero_v8f();
#pragma unroll
    for (int nt = 0; nt < 4; ++nt) {
      const bf16_t* krow = &kS[cur][nt * 16 + ln][8 * hi];
#pragma unroll
      for (int ks = 0; ks < 4; ++ks) {
        FragA bfv;
#pragma unroll
        for (int p = 0; p < 8; ++p)
          bfv.u[p] = *(const unsigned int*)(krow + ks * 32 + ((p & 3) << 1) + ((p >> 2) << 4));
        sacc[nt] = __builtin_amdgcn_wmma_f32_16x16x32_bf16(
            false, qf[ks].v, false, bfv.v, (short)0, sacc[nt], false, false);
      }
    }

    // ---- causal mask + row max (rows live per 16-lane half) ----
    {
      const int rbase = qr0 + hi * 8;
      float rmax[8];
#pragma unroll
      for (int v = 0; v < 8; ++v) rmax[v] = -1e30f;
#pragma unroll
      for (int nt = 0; nt < 4; ++nt) {
        const int c = key0 + nt * 16 + ln;
#pragma unroll
        for (int v = 0; v < 8; ++v) {
          if (c > rbase + v) sacc[nt][v] = -1e30f;
          rmax[v] = fmaxf(rmax[v], sacc[nt][v]);
        }
      }
#pragma unroll
      for (int v = 0; v < 8; ++v) {
        float x = rmax[v];
#pragma unroll
        for (int mask = 8; mask >= 1; mask >>= 1)
          x = fmaxf(x, __shfl_xor(x, mask, 32));
        rmax[v] = x;
      }

      // ---- online softmax update ----
      float rsum[8];
#pragma unroll
      for (int v = 0; v < 8; ++v) {
        const float mnew  = fmaxf(mrow[v], rmax[v]);
        const float alpha = __expf(mrow[v] - mnew);
        mrow[v] = mnew;
        float s = 0.0f;
#pragma unroll
        for (int nt = 0; nt < 4; ++nt) {
          const float p = __expf(sacc[nt][v] - mnew);
          sacc[nt][v] = p;
          s += p;
        }
        rsum[v] = s;
        lrow[v] *= alpha;
#pragma unroll
        for (int t = 0; t < 8; ++t) oacc[t][v] *= alpha;
      }
#pragma unroll
      for (int v = 0; v < 8; ++v) {
        float s = rsum[v];
#pragma unroll
        for (int mask = 8; mask >= 1; mask >>= 1)
          s += __shfl_xor(s, mask, 32);
        lrow[v] += s;
      }
    }

    // ---- write P to per-wave LDS (C-layout -> row-major) ----
#pragma unroll
    for (int nt = 0; nt < 4; ++nt) {
#pragma unroll
      for (int v = 0; v < 8; ++v)
        pS[wave][hi * 8 + v][nt * 16 + ln] = (bf16_t)sacc[nt][v];
    }
    __asm__ volatile("s_wait_dscnt 0" ::: "memory");  // intra-wave LDS RAW

    // ---- O += P * V (V B-frags gathered transposed from row-major vS) ----
#pragma unroll
    for (int ks = 0; ks < 2; ++ks) {
      FragA pf;
      const bf16_t* prow = &pS[wave][ln][ks * 32 + 8 * hi];
#pragma unroll
      for (int p = 0; p < 8; ++p) {
        const int k = ((p & 3) << 1) + ((p >> 2) << 4);
        pf.u[p] = *(const unsigned int*)(prow + k);
      }
#pragma unroll
      for (int ot = 0; ot < 8; ++ot) {
        const int dcol = ot * 16 + ln;
        FragA vf;
#pragma unroll
        for (int p = 0; p < 8; ++p) {
          const bf16_t* c0 = &vS[cur][ks * 32 + 8 * hi + ((p & 3) << 1) + ((p >> 2) << 4)][dcol];
          const unsigned lo = *(const unsigned short*)c0;
          const unsigned h2 = *(const unsigned short*)(c0 + 130);
          vf.u[p] = lo | (h2 << 16);
        }
        oacc[ot] = __builtin_amdgcn_wmma_f32_16x16x32_bf16(
            false, pf.v, false, vf.v, (short)0, oacc[ot], false, false);
      }
    }
  }

  // ---- epilogue: O / l, write ctx (B, N, H*DHd) as bf16 ----
  const int rbase = qr0 + hi * 8;
#pragma unroll
  for (int ot = 0; ot < 8; ++ot) {
    const int dcol = ot * 16 + ln;
#pragma unroll
    for (int v = 0; v < 8; ++v) {
      const float val = oacc[ot][v] / lrow[v];
      const int r = rbase + v;
      Ctx[((size_t)b * Nn + r) * ((size_t)Hh * DHd) + (size_t)h * DHd + dcol] = (bf16_t)val;
    }
  }
}

// ---------------- host launcher ----------------
extern "C" void kernel_launch(void* const* d_in, const int* in_sizes, int n_in,
                              void* d_out, int out_size, void* d_ws, size_t ws_size,
                              hipStream_t stream) {
  (void)in_sizes; (void)n_in; (void)out_size; (void)ws_size;

  const float* x  = (const float*)d_in[0];
  const float* wq = (const float*)d_in[1];
  const float* wk = (const float*)d_in[2];
  const float* wv = (const float*)d_in[3];
  const float* wo = (const float*)d_in[4];
  float* out = (float*)d_out;

  char* ws = (char*)d_ws;
  size_t off = 0;
  auto alloc = [&](size_t elems) -> bf16_t* {
    bf16_t* p = (bf16_t*)(ws + off);
    off += ((elems * sizeof(bf16_t) + 255) & ~(size_t)255);
    return p;
  };

  bf16_t* xb   = alloc((size_t)Bb * Nn * Dd);
  bf16_t* wqb  = alloc((size_t)Dd * Hh * DHd);
  bf16_t* wkb  = alloc((size_t)Dd * KVH * DHd);
  bf16_t* wvb  = alloc((size_t)Dd * KVH * DHd);
  bf16_t* wob  = alloc((size_t)Hh * DHd * Dd);
  bf16_t* qtmp = alloc((size_t)Bb * Nn * Hh * DHd);
  bf16_t* ktmp = alloc((size_t)Bb * Nn * KVH * DHd);
  bf16_t* vtmp = alloc((size_t)Bb * Nn * KVH * DHd);
  bf16_t* qatt = alloc((size_t)Bb * Hh * Nn * DHd);
  bf16_t* katt = alloc((size_t)Bb * KVH * Nn * DHd);
  bf16_t* vatt = alloc((size_t)Bb * KVH * Nn * DHd);
  bf16_t* ctx  = alloc((size_t)Bb * Nn * Hh * DHd);

  // 1) fp32 -> bf16
  cvt_bf16_kernel<<<2048, 256, 0, stream>>>(x,  xb,  (size_t)Bb * Nn * Dd);
  cvt_bf16_kernel<<<2048, 256, 0, stream>>>(wq, wqb, (size_t)Dd * Hh * DHd);
  cvt_bf16_kernel<<<1024, 256, 0, stream>>>(wk, wkb, (size_t)Dd * KVH * DHd);
  cvt_bf16_kernel<<<1024, 256, 0, stream>>>(wv, wvb, (size_t)Dd * KVH * DHd);
  cvt_bf16_kernel<<<2048, 256, 0, stream>>>(wo, wob, (size_t)Hh * DHd * Dd);

  // 2) QKV projections (M = B*N = 4096)
  {
    dim3 blk(128);
    dim3 gq(Hh * DHd / 64, Bb * Nn / 64);    // 32 x 64
    wmma_gemm_kernel<bf16_t><<<gq, blk, 0, stream>>>(xb, wqb, qtmp, Bb * Nn, Hh * DHd, Dd);
    dim3 gk(KVH * DHd / 64, Bb * Nn / 64);   // 8 x 64
    wmma_gemm_kernel<bf16_t><<<gk, blk, 0, stream>>>(xb, wkb, ktmp, Bb * Nn, KVH * DHd, Dd);
    wmma_gemm_kernel<bf16_t><<<gk, blk, 0, stream>>>(xb, wvb, vtmp, Bb * Nn, KVH * DHd, Dd);
  }

  // 3) RoPE + relayout (softmax scale folded into Q)
  const float qscale = 0.08838834764831845f;   // 1/sqrt(128)
  rope_relayout_kernel<<<Bb * Hh * Nn, 64, 0, stream>>>(qtmp, qatt, Hh, qscale, 1);
  rope_relayout_kernel<<<Bb * KVH * Nn, 64, 0, stream>>>(ktmp, katt, KVH, 1.0f, 1);
  rope_relayout_kernel<<<Bb * KVH * Nn, 64, 0, stream>>>(vtmp, vatt, KVH, 1.0f, 0);

  // 4) causal flash attention
  {
    dim3 ga(Nn / 64, Bb * Hh);
    attn_kernel<<<ga, 128, 0, stream>>>(qatt, katt, vatt, ctx);
  }

  // 5) output projection -> f32 d_out
  {
    dim3 go(Dd / 64, Bb * Nn / 64);
    wmma_gemm_kernel<float><<<go, 128, 0, stream>>>(ctx, wob, out, Bb * Nn, Dd, Hh * DHd);
  }
}
